// PointCloudDecoder_42417097015901
// MI455X (gfx1250) — matrix operand order repaired and checked by reading, not compiled, not run on hardware
//
#include <hip/hip_runtime.h>
#include <hip/hip_bf16.h>

typedef __attribute__((ext_vector_type(16))) _Float16 v16h;
typedef __attribute__((ext_vector_type(8)))  float    v8f;

#define BB 4
#define FF 128
#define PP 1024
#define DD 512
#define GG 16
#define D4 128
#define HH 8
#define HDD 64
#define LL 8

__device__ __forceinline__ float gelu_tanh(float x) {
    // jax.nn.gelu default (approximate=True)
    const float c0 = 0.7978845608028654f; // sqrt(2/pi)
    const float c1 = 0.044715f;
    float x3 = x * x * x;
    float t = tanhf(c0 * (x + c1 * x3));
    return 0.5f * x * (1.0f + t);
}

// ---------------------------------------------------------------------------
// WMMA GEMM: C[M,N] = epi(A[M,K] @ W[K,N] + bias [+ R])
// EPI: 0 = store, 1 = gelu then store, 2 = add residual R then store
// Block = 256 threads (8 waves) computes a 128x128 C tile:
//   wave w owns rows m0 = mb*128 + w*16, all 128 cols (8 accumulators).
// W k-tile (32x128) is staged in LDS as f16 [n][k] (row stride 40 halves =
// 80 B, 16B-aligned) so each B fragment is one contiguous 32-byte LDS read.
// Requires M%128==0, N%128==0, K%32==0.
// ---------------------------------------------------------------------------
__global__ __launch_bounds__(256) void gemm_wmma_kernel(
    const float* __restrict__ A, const float* __restrict__ W,
    const float* __restrict__ bias, const float* __restrict__ R,
    float* __restrict__ C, int M, int N, int K, int EPI)
{
    __shared__ _Float16 wtile[128 * 40];   // 10 KB

    int tid  = threadIdx.x;
    int lane = tid & 31;
    int wave = tid >> 5;
    int nblk = N >> 7;
    int nb   = blockIdx.x % nblk;
    int mb   = blockIdx.x / nblk;
    int m0   = mb * 128 + wave * 16;
    int n0   = nb * 128;
    int lm   = lane & 15;
    int half = lane >> 4;
    int mrow = m0 + lm;

    v8f accs[8];
    #pragma unroll
    for (int i = 0; i < 8; ++i) accs[i] = (v8f){};

    for (int k0 = 0; k0 < K; k0 += 32) {
        // ---- cooperative W tile load: 32 k-rows x 128 n, float4 coalesced ----
        __syncthreads();   // previous iteration's B reads complete
        #pragma unroll
        for (int it = 0; it < 4; ++it) {
            int vec = tid + it * 256;          // 0..1023
            int kk  = vec >> 5;                // 0..31
            int n4  = (vec & 31) << 2;         // 0..124
            const float4 wv = *(const float4*)(W + (size_t)(k0 + kk) * N + n0 + n4);
            wtile[(n4 + 0) * 40 + kk] = (_Float16)wv.x;
            wtile[(n4 + 1) * 40 + kk] = (_Float16)wv.y;
            wtile[(n4 + 2) * 40 + kk] = (_Float16)wv.z;
            wtile[(n4 + 3) * 40 + kk] = (_Float16)wv.w;
        }
        __syncthreads();

        // ---- A fragment (16 rows x 32 k), reused across 8 n-tiles ----
        v16h a;
        const float* Arow = A + (size_t)mrow * K + k0;
        #pragma unroll
        for (int j = 0; j < 8; ++j) {
            int kb = half * 16 + j * 2;
            a[2*j]   = (_Float16)Arow[kb];
            a[2*j+1] = (_Float16)Arow[kb + 1];
        }

        #pragma unroll
        for (int nt = 0; nt < 8; ++nt) {
            int n = nt * 16 + lm;
            const v16h bf = *(const v16h*)&wtile[n * 40 + half * 16];
            accs[nt] = __builtin_amdgcn_wmma_f32_16x16x32_f16(false, a, false, bf,
                                                              (short)0, accs[nt],
                                                              false, false);
        }
    }

    // ---- epilogue ----
    #pragma unroll
    for (int nt = 0; nt < 8; ++nt) {
        int n = n0 + nt * 16 + lm;
        float bn = bias ? bias[n] : 0.0f;
        #pragma unroll
        for (int r = 0; r < 8; ++r) {
            int m = m0 + r + half * 8;
            float v = accs[nt][r] + bn;
            if (EPI == 1)      v = gelu_tanh(v);
            else if (EPI == 2) v += R[(size_t)m * N + n];
            C[(size_t)m * N + n] = v;
        }
    }
}

// ---------------------------------------------------------------------------
// RMSNorm: Y[row] = X[row] * w * rsqrt(mean(X^2)+eps)
// ---------------------------------------------------------------------------
__global__ __launch_bounds__(256) void rmsnorm_kernel(
    const float* __restrict__ X, const float* __restrict__ w,
    float* __restrict__ Y, int D)
{
    int row = blockIdx.x;
    const float* x = X + (size_t)row * D;
    float ss = 0.0f;
    for (int i = threadIdx.x; i < D; i += blockDim.x) { float v = x[i]; ss += v * v; }
    __shared__ float red[32];
    for (int off = 16; off > 0; off >>= 1) ss += __shfl_down(ss, off, 32);
    int lane = threadIdx.x & 31, wid = threadIdx.x >> 5;
    if (lane == 0) red[wid] = ss;
    __syncthreads();
    int nw = blockDim.x >> 5;
    if (wid == 0) {
        float v = (lane < nw) ? red[lane] : 0.0f;
        for (int off = 16; off > 0; off >>= 1) v += __shfl_down(v, off, 32);
        if (lane == 0) red[0] = v;
    }
    __syncthreads();
    float inv = rsqrtf(red[0] / (float)D + 1e-6f);
    for (int i = threadIdx.x; i < D; i += blockDim.x)
        Y[(size_t)row * D + i] = x[i] * w[i] * inv;
}

// ---------------------------------------------------------------------------
// RoPE in-place on (B,F,D) buffer, per head of HD=64 (halves of 32).
// ---------------------------------------------------------------------------
__global__ __launch_bounds__(256) void rope_kernel(float* __restrict__ Q)
{
    int idx = blockIdx.x * blockDim.x + threadIdx.x;   // B*F*H*32 = 131072
    if (idx >= BB * FF * HH * 32) return;
    int i = idx % 32; int t = idx / 32;
    int h = t % HH;   t /= HH;
    int f = t % FF;   int b = t / FF;
    float inv = powf(10000.0f, -(2.0f * (float)i) / 64.0f);
    float ang = (float)f * inv;
    float c = cosf(ang), s = sinf(ang);
    size_t base = ((size_t)(b * FF + f)) * DD + h * HDD + i;
    float x1 = Q[base], x2 = Q[base + 32];
    Q[base]      = x1 * c - x2 * s;
    Q[base + 32] = x2 * c + x1 * s;
}

// ---------------------------------------------------------------------------
// Exact fp32 attention, one block per (b,h), 128 threads = 1 query row each.
// Online softmax, K/V staged in LDS (64 KB).
// ---------------------------------------------------------------------------
__global__ __launch_bounds__(128) void attn_kernel(
    const float* __restrict__ Q, const float* __restrict__ Kb,
    const float* __restrict__ Vb, float* __restrict__ O)
{
    __shared__ float ks[FF][HDD];
    __shared__ float vs[FF][HDD];
    int bh = blockIdx.x; int b = bh >> 3; int h = bh & 7;
    int t = threadIdx.x;
    const float* krow = Kb + ((size_t)(b * FF + t)) * DD + h * HDD;
    const float* vrow = Vb + ((size_t)(b * FF + t)) * DD + h * HDD;
    for (int d = 0; d < HDD; ++d) { ks[t][d] = krow[d]; vs[t][d] = vrow[d]; }
    __syncthreads();
    float q[HDD], o[HDD];
    const float* qrow = Q + ((size_t)(b * FF + t)) * DD + h * HDD;
    for (int d = 0; d < HDD; ++d) { q[d] = qrow[d]; o[d] = 0.0f; }
    float m = -1e30f, ssum = 0.0f;
    const float scale = 0.125f; // 1/sqrt(64)
    for (int j = 0; j < FF; ++j) {
        float s = 0.0f;
        for (int d = 0; d < HDD; ++d) s += q[d] * ks[j][d];
        s *= scale;
        float mn   = fmaxf(m, s);
        float corr = expf(m - mn);
        float p    = expf(s - mn);
        ssum = ssum * corr + p;
        for (int d = 0; d < HDD; ++d) o[d] = o[d] * corr + p * vs[j][d];
        m = mn;
    }
    float inv = 1.0f / ssum;
    float* orow = O + ((size_t)(b * FF + t)) * DD + h * HDD;
    for (int d = 0; d < HDD; ++d) orow[d] = o[d] * inv;
}

// ---------------------------------------------------------------------------
// idx[b,p] = argmax_g bgroups[b,p,g]
// ---------------------------------------------------------------------------
__global__ void argmax_kernel(const float* __restrict__ bg, int* __restrict__ idx)
{
    int i = blockIdx.x * blockDim.x + threadIdx.x;
    if (i >= BB * PP) return;
    const float* r = bg + (size_t)i * GG;
    int best = 0; float bv = r[0];
    for (int g = 1; g < GG; ++g) if (r[g] > bv) { bv = r[g]; best = g; }
    idx[i] = best;
}

// ---------------------------------------------------------------------------
// Build padded concat input (512 x 544) and padded embed_in_w (544 x 512)
// ---------------------------------------------------------------------------
#define KPAD 544
__global__ void build_a0(const float* __restrict__ emb, const float* __restrict__ offs,
                         float* __restrict__ A0)
{
    int i = blockIdx.x * blockDim.x + threadIdx.x;
    if (i >= BB * FF * KPAD) return;
    int m = i / KPAD, k = i % KPAD;
    float v = 0.0f;
    if (k < DD)            v = emb[(size_t)m * DD + k];
    else if (k < DD + 3)   v = offs[(size_t)m * 3 + (k - DD)];
    A0[i] = v;
}
__global__ void build_wpad(const float* __restrict__ w, float* __restrict__ Wp)
{
    int i = blockIdx.x * blockDim.x + threadIdx.x;
    if (i >= KPAD * DD) return;
    int k = i / DD, n = i % DD;
    Wp[i] = (k < DD + 3) ? w[(size_t)k * DD + n] : 0.0f;
}

// ---------------------------------------------------------------------------
// Fused point stage: per block = one (b,f,ptile of 128 points).
// z (128x128 fp32) lives in LDS; 3 residual gelu-MLP layers via WMMA,
// then output head + bgroup broadcast. 8 waves, each wave owns 16 rows of z.
// ---------------------------------------------------------------------------
__global__ __launch_bounds__(256) void point_kernel(
    const float* __restrict__ xbuf,    // (B,F,G,128)
    const float* __restrict__ seedp,   // (B,P,128)
    const int*   __restrict__ idx,     // (B,P)
    const float* __restrict__ offs,    // (B,F,3)
    const float* __restrict__ bgroups, // (B,P,16)
    const float* __restrict__ w1, const float* __restrict__ b1,
    const float* __restrict__ w2, const float* __restrict__ b2,
    const float* __restrict__ w3, const float* __restrict__ b3,
    const float* __restrict__ wo, const float* __restrict__ bo,
    float* __restrict__ out)
{
    __shared__ float z[128 * 128];   // 64 KB
    int blk = blockIdx.x;
    int pt = blk & 7; int f = (blk >> 3) & (FF - 1); int b = blk >> 10;
    int p0 = pt * 128;
    int tid = threadIdx.x;

    // z0 = x[b,f,idx[b,p],:] + seed_p[b,p,:]
    for (int i = tid; i < 128 * 128; i += 256) {
        int m = i >> 7, k = i & 127;
        int p = p0 + m;
        int g = idx[b * PP + p];
        z[i] = xbuf[(size_t)((b * FF + f) * GG + g) * D4 + k]
             + seedp[(size_t)(b * PP + p) * D4 + k];
    }
    __syncthreads();

    int lane = tid & 31, wave = tid >> 5;
    int m0 = wave * 16;
    int half = lane >> 4;
    int lm = lane & 15;

    const float* Ws[3] = { w1, w2, w3 };
    const float* Bs[3] = { b1, b2, b3 };
    for (int layer = 0; layer < 3; ++layer) {
        const float* W = Ws[layer];
        const float* bias = Bs[layer];
        // A fragments = gelu(z rows m0..m0+15), all K (4 k-tiles of 32)
        v16h afr[4];
        #pragma unroll
        for (int kt = 0; kt < 4; ++kt) {
            int k0 = kt * 32;
            #pragma unroll
            for (int j = 0; j < 8; ++j) {
                int kb = k0 + half * 16 + j * 2;
                afr[kt][2*j]   = (_Float16)gelu_tanh(z[(m0 + lm) * 128 + kb]);
                afr[kt][2*j+1] = (_Float16)gelu_tanh(z[(m0 + lm) * 128 + kb + 1]);
            }
        }
        v8f accs[8];
        #pragma unroll
        for (int ntile = 0; ntile < 8; ++ntile) {
            int n = ntile * 16 + lm;
            v8f acc = {};
            #pragma unroll
            for (int kt = 0; kt < 4; ++kt) {
                v16h bf;
                int kbase = kt * 32 + half * 16;
                #pragma unroll
                for (int j = 0; j < 8; ++j) {
                    bf[2*j]   = (_Float16)W[(size_t)(kbase + 2*j) * D4 + n];
                    bf[2*j+1] = (_Float16)W[(size_t)(kbase + 2*j + 1) * D4 + n];
                }
                acc = __builtin_amdgcn_wmma_f32_16x16x32_f16(false, afr[kt], false, bf,
                                                             (short)0, acc, false, false);
            }
            accs[ntile] = acc;
        }
        // z += acc + bias  (wave-private 16-row band)
        #pragma unroll
        for (int ntile = 0; ntile < 8; ++ntile) {
            int n = ntile * 16 + lm;
            float bn = bias[n];
            #pragma unroll
            for (int r = 0; r < 8; ++r) {
                int m = m0 + r + half * 8;
                z[m * 128 + n] += accs[ntile][r] + bn;
            }
        }
        __syncthreads();   // ordering safety between layers
    }

    // output head: out[...,0:3] = z @ outp_w + outp_b + offsets; out[...,3:19] = bgroups
    if (tid < 128) {
        int p = p0 + tid;
        float o0 = bo[0], o1 = bo[1], o2 = bo[2];
        for (int k = 0; k < 128; ++k) {
            float zv = z[tid * 128 + k];
            o0 += zv * wo[k * 3 + 0];
            o1 += zv * wo[k * 3 + 1];
            o2 += zv * wo[k * 3 + 2];
        }
        const float* off = offs + (size_t)(b * FF + f) * 3;
        float* orow = out + (size_t)((size_t)(b * FF + f) * PP + p) * 19;
        orow[0] = o0 + off[0];
        orow[1] = o1 + off[1];
        orow[2] = o2 + off[2];
    } else {
        int p = p0 + (tid - 128);
        const float* bgr = bgroups + (size_t)(b * PP + p) * GG;
        float* orow = out + (size_t)((size_t)(b * FF + f) * PP + p) * 19 + 3;
        for (int g = 0; g < GG; ++g) orow[g] = bgr[g];
    }
}

// ---------------------------------------------------------------------------
extern "C" void kernel_launch(void* const* d_in, const int* in_sizes, int n_in,
                              void* d_out, int out_size, void* d_ws, size_t ws_size,
                              hipStream_t stream)
{
    (void)in_sizes; (void)n_in; (void)out_size; (void)ws_size;
    const float* embeds     = (const float*)d_in[0];
    const float* offsets    = (const float*)d_in[1];
    const float* bgroups    = (const float*)d_in[2];
    const float* seed       = (const float*)d_in[3];
    const float* embed_in_w = (const float*)d_in[4];
    const float* embed_in_b = (const float*)d_in[5];
    const float* norm1_w    = (const float*)d_in[6];
    const float* wq         = (const float*)d_in[7];
    const float* bq         = (const float*)d_in[8];
    const float* wk         = (const float*)d_in[9];
    const float* bk         = (const float*)d_in[10];
    const float* wv         = (const float*)d_in[11];
    const float* bv         = (const float*)d_in[12];
    const float* wo         = (const float*)d_in[13];
    const float* bo         = (const float*)d_in[14];
    const float* norm2_w    = (const float*)d_in[15];
    const float* ffn_w1     = (const float*)d_in[16];
    const float* ffn_b1     = (const float*)d_in[17];
    const float* ffn_w2     = (const float*)d_in[18];
    const float* ffn_b2     = (const float*)d_in[19];
    const float* bgroup_w   = (const float*)d_in[20];
    const float* bgroup_b   = (const float*)d_in[21];
    const float* point_w    = (const float*)d_in[22];
    const float* point_b    = (const float*)d_in[23];
    const float* mlp1_w     = (const float*)d_in[24];
    const float* mlp1_b     = (const float*)d_in[25];
    const float* mlp2_w     = (const float*)d_in[26];
    const float* mlp2_b     = (const float*)d_in[27];
    const float* mlp3_w     = (const float*)d_in[28];
    const float* mlp3_b     = (const float*)d_in[29];
    const float* outp_w     = (const float*)d_in[30];
    const float* outp_b     = (const float*)d_in[31];
    float* outp = (float*)d_out;

    const int ROWS = BB * FF;                // 512
    float* ws = (float*)d_ws;
    size_t off = 0;
    float* A0 = ws + off;  off += (size_t)ROWS * KPAD;   // 512x544
    float* Wp = ws + off;  off += (size_t)KPAD * DD;     // 544x512
    float* h  = ws + off;  off += (size_t)ROWS * DD;
    float* y  = ws + off;  off += (size_t)ROWS * DD;
    float* q  = ws + off;  off += (size_t)ROWS * DD;
    float* k  = ws + off;  off += (size_t)ROWS * DD;
    float* v  = ws + off;  off += (size_t)ROWS * DD;
    float* o  = ws + off;  off += (size_t)ROWS * DD;
    float* t  = ws + off;  off += (size_t)ROWS * 4 * DD; // 512x2048
    float* xb = ws + off;  off += (size_t)ROWS * 4 * DD; // 512x2048
    float* sp = ws + off;  off += (size_t)BB * PP * D4;  // 4096x128
    int*   gi = (int*)(ws + off);                        // 4096 ints

    auto gemm = [&](const float* A, const float* W, const float* bias, const float* R,
                    float* C, int M, int N, int K, int EPI) {
        int blocks = (M / 128) * (N / 128);
        gemm_wmma_kernel<<<blocks, 256, 0, stream>>>(A, W, bias, R, C, M, N, K, EPI);
    };

    // Prep
    build_a0<<<(ROWS * KPAD + 255) / 256, 256, 0, stream>>>(embeds, offsets, A0);
    build_wpad<<<(KPAD * DD + 255) / 256, 256, 0, stream>>>(embed_in_w, Wp);
    argmax_kernel<<<(BB * PP + 255) / 256, 256, 0, stream>>>(bgroups, gi);

    // h = concat(embeds, offsets) @ embed_in_w + b
    gemm(A0, Wp, embed_in_b, nullptr, h, ROWS, DD, KPAD, 0);
    // seed_p = seed @ point_w + point_b
    gemm(seed, point_w, point_b, nullptr, sp, BB * PP, D4, D4, 0);

    for (int l = 0; l < LL; ++l) {
        const size_t wsz = (size_t)DD * DD;
        const size_t f1  = (size_t)DD * 4 * DD;
        rmsnorm_kernel<<<ROWS, 256, 0, stream>>>(h, norm1_w + (size_t)l * DD, y, DD);
        gemm(y, wq + l * wsz, bq + (size_t)l * DD, nullptr, q, ROWS, DD, DD, 0);
        gemm(y, wk + l * wsz, bk + (size_t)l * DD, nullptr, k, ROWS, DD, DD, 0);
        gemm(y, wv + l * wsz, bv + (size_t)l * DD, nullptr, v, ROWS, DD, DD, 0);
        rope_kernel<<<(BB * FF * HH * 32 + 255) / 256, 256, 0, stream>>>(q);
        rope_kernel<<<(BB * FF * HH * 32 + 255) / 256, 256, 0, stream>>>(k);
        attn_kernel<<<BB * HH, 128, 0, stream>>>(q, k, v, o);
        gemm(o, wo + l * wsz, bo + (size_t)l * DD, h, h, ROWS, DD, DD, 2);
        rmsnorm_kernel<<<ROWS, 256, 0, stream>>>(h, norm2_w + (size_t)l * DD, y, DD);
        gemm(y, ffn_w1 + l * f1, ffn_b1 + (size_t)l * 4 * DD, nullptr, t, ROWS, 4 * DD, DD, 1);
        gemm(t, ffn_w2 + l * f1, ffn_b2 + (size_t)l * DD, h, h, ROWS, DD, 4 * DD, 2);
    }

    // x = h @ bgroup_w + b   -> (B,F,G,D4)
    gemm(h, bgroup_w, bgroup_b, nullptr, xb, ROWS, GG * D4, DD, 0);

    // Fused point stage
    point_kernel<<<BB * FF * (PP / 128), 256, 0, stream>>>(
        xb, sp, gi, offsets, bgroups,
        mlp1_w, mlp1_b, mlp2_w, mlp2_b, mlp3_w, mlp3_b,
        outp_w, outp_b, outp);
}
